// CausalSelfAttention_16149077032974
// MI455X (gfx1250) — matrix operand order repaired and checked by reading, not compiled, not run on hardware
//
#include <hip/hip_runtime.h>
#include <hip/hip_bf16.h>
#include <math.h>

typedef __attribute__((ext_vector_type(16))) __bf16       v16bf;
typedef __attribute__((ext_vector_type(8)))  float        v8f;
typedef __attribute__((ext_vector_type(4)))  unsigned int u32x4;

#define B_  2
#define T_  2048
#define H_  2048
#define NH_ 16
#define HD_ 128
#define MT_ (B_*T_)   /* 4096 rows */

// ---------------------------------------------------------------------------
// WMMA helper: D = A(16x32 bf16) * B(32x16 bf16) + C(16x16 f32)
// ---------------------------------------------------------------------------
static __device__ __forceinline__ v8f wmma_bf16(v16bf a, v16bf b, v8f c) {
  return __builtin_amdgcn_wmma_f32_16x16x32_bf16(false, a, false, b, (short)0, c,
                                                 false, false);
}

// A-fragment (16x32, 16-bit): lane holds row m=lane&15, K range
// [c,c+16) with c = 16*(lane>=16). Caller passes p already at
// (row base + k0 + c); 32 contiguous bytes at p.
static __device__ __forceinline__ void load_frag_A(v16bf& f, const __bf16* p) {
  u32x4* q = reinterpret_cast<u32x4*>(&f);
  q[0] = *reinterpret_cast<const u32x4*>(p);
  q[1] = *reinterpret_cast<const u32x4*>(p + 8);
}
// B-fragment (32x16, 16-bit): lane holds column n=lane&15, K range
// [16*(lane>=16), +16) -> 32 contiguous bytes at p.
static __device__ __forceinline__ void load_frag_B(v16bf& f, const __bf16* p) {
  u32x4* q = reinterpret_cast<u32x4*>(&f);
  q[0] = *reinterpret_cast<const u32x4*>(p);
  q[1] = *reinterpret_cast<const u32x4*>(p + 8);
}

// ---------------------------------------------------------------------------
// C[4096,2048] f32 = A[4096,2048] bf16 (row-major) * W[2048,2048] bf16 (W^T).
// Dimensions are compile-time so all fragment/store addressing folds to
// base + immediate offsets. block = 256 threads (8 waves), tile 128x128,
// wave tile 32x64 (2 M-frags x 4 N-frags, 8 accumulators).
// ---------------------------------------------------------------------------
__global__ __launch_bounds__(256) void gemm_bf16_nt(const __bf16* __restrict__ A,
                                                    const __bf16* __restrict__ W,
                                                    float* __restrict__ C)
{
  constexpr int N = H_;
  constexpr int K = H_;

  const int lane = threadIdx.x & 31;
  const int w    = threadIdx.x >> 5;
  const int mw   = w & 3;      // 4 M-subtiles of 32 rows
  const int nw   = w >> 2;     // 2 N-subtiles of 64 cols
  const int col  = lane & 15;
  const int hi   = lane >> 4;
  const int rowA = blockIdx.y * 128 + mw * 32;
  const int colB = blockIdx.x * 128 + nw * 64;

  const v8f vz = {0.f,0.f,0.f,0.f,0.f,0.f,0.f,0.f};
  v8f acc[2][4];
#pragma unroll
  for (int mi = 0; mi < 2; ++mi)
#pragma unroll
    for (int ni = 0; ni < 4; ++ni) acc[mi][ni] = vz;

  // Per-lane row base pointers (constant across the K loop).
  const __bf16* pa0 = A + (size_t)(rowA + col) * K + hi * 16;      // mi stride 16*K
  const __bf16* pb0 = W + (size_t)(colB + col) * K + hi * 16;      // ni stride 16*K

  for (int k0 = 0; k0 < K; k0 += 32) {
    v16bf af[2];
#pragma unroll
    for (int mi = 0; mi < 2; ++mi)
      load_frag_A(af[mi], pa0 + (size_t)(16 * mi) * K + k0);
    v16bf bf[4];
#pragma unroll
    for (int ni = 0; ni < 4; ++ni)
      load_frag_B(bf[ni], pb0 + (size_t)(16 * ni) * K + k0);

    // Warm the next K tile into cache (global_prefetch_b8, no LOADcnt).
    if (k0 + 32 < K) {
      __builtin_prefetch(pa0 + k0 + 32, 0, 0);
      __builtin_prefetch(pb0 + k0 + 32, 0, 0);
    }

#pragma unroll
    for (int mi = 0; mi < 2; ++mi)
#pragma unroll
      for (int ni = 0; ni < 4; ++ni)
        acc[mi][ni] = wmma_bf16(af[mi], bf[ni], acc[mi][ni]);
  }

  // C/D layout: lane holds column n=lane&15, VGPR r -> row m = r + 8*(lane>=16).
  // One base per mi; everything else is an immediate offset (r*N*4 + ni*64 B).
#pragma unroll
  for (int mi = 0; mi < 2; ++mi) {
    float* cp = C + (size_t)(rowA + 16*mi + 8*hi) * N + colB + col;
#pragma unroll
    for (int ni = 0; ni < 4; ++ni)
#pragma unroll
      for (int r = 0; r < 8; ++r)
        cp[(size_t)r * N + 16 * ni] = acc[mi][ni][r];
  }
}

// ---------------------------------------------------------------------------
// elementwise f32 -> bf16
// ---------------------------------------------------------------------------
__global__ void f32_to_bf16_k(const float* __restrict__ s, __bf16* __restrict__ d,
                              long n) {
  long i = (long)blockIdx.x * blockDim.x + threadIdx.x;
  long st = (long)gridDim.x * blockDim.x;
  for (; i < n; i += st) d[i] = (__bf16)s[i];
}

// ---------------------------------------------------------------------------
// RoPE + per-head RMS-norm. One wave per (b,t,h) 128-vector.
// src layout [B,T,NH,HD] (f32), dst layout [B,NH,T,HD] (bf16).
// Lane holds d, d+32, d+64, d+96 so rope pairs (d, d+64) stay in-lane.
// ---------------------------------------------------------------------------
__global__ __launch_bounds__(32) void rope_rms_k(const float* __restrict__ src,
                                                 const float* __restrict__ gamma,
                                                 __bf16* __restrict__ dst)
{
  const int id = blockIdx.x;            // id = (b*T + t)*NH + h
  const int h  = id % NH_;
  const int t  = (id / NH_) % T_;
  const int b  = id / (NH_ * T_);
  const int lane = threadIdx.x;

  const float* x = src + (size_t)id * HD_;
  float x0 = x[lane], x1 = x[lane + 32], x2 = x[lane + 64], x3 = x[lane + 96];

  const float LOG1E4 = 9.210340371976184f;      // ln(10000)
  float a0 = (float)t * __expf(-LOG1E4 * (float)lane        * (1.0f/64.0f));
  float a1 = (float)t * __expf(-LOG1E4 * (float)(lane + 32) * (1.0f/64.0f));
  float c0 = __cosf(a0), s0 = __sinf(a0);
  float c1 = __cosf(a1), s1 = __sinf(a1);
  float r0 = x0*c0 - x2*s0;   // d < 64  : x*cos - x[d+64]*sin
  float r2 = x2*c0 + x0*s0;   // d >= 64 : x*cos + x[d-64]*sin
  float r1 = x1*c1 - x3*s1;
  float r3 = x3*c1 + x1*s1;

  float ss = r0*r0 + r1*r1 + r2*r2 + r3*r3;
#pragma unroll
  for (int m = 1; m < 32; m <<= 1) ss += __shfl_xor(ss, m, 32);
  float rinv = rsqrtf(ss * (1.0f/128.0f) + 1e-5f);

  __bf16* o = dst + (((size_t)(b*NH_ + h)) * T_ + t) * HD_;
  o[lane]      = (__bf16)(r0 * rinv * gamma[lane]);
  o[lane + 32] = (__bf16)(r1 * rinv * gamma[lane + 32]);
  o[lane + 64] = (__bf16)(r2 * rinv * gamma[lane + 64]);
  o[lane + 96] = (__bf16)(r3 * rinv * gamma[lane + 96]);
}

// ---------------------------------------------------------------------------
// V f32 [B,T,NH,HD] -> Vt bf16 [B,NH,HD,T] (t-contiguous for PV B-operand)
// ---------------------------------------------------------------------------
__global__ void v_transpose_k(const float* __restrict__ V, __bf16* __restrict__ Vt,
                              long n) {
  long i = (long)blockIdx.x * blockDim.x + threadIdx.x;
  long st = (long)gridDim.x * blockDim.x;
  for (; i < n; i += st) {
    int  t  = (int)(i % T_);
    long r  = i / T_;              // r = (b*NH + h)*HD + d
    int  d  = (int)(r % HD_);
    int  bh = (int)(r / HD_);
    int  b  = bh / NH_, h = bh % NH_;
    Vt[i] = (__bf16)V[(((size_t)(b*T_ + t)) * NH_ + h) * HD_ + d];
  }
}

// ---------------------------------------------------------------------------
// Flash attention + o-RMS + SiLU gate. One wave (block of 32) per 16 q rows.
// Qb,Kb: [B,NH,T,HD] bf16; Vt: [B,NH,HD,T] bf16; Gf: [B,T,NH,HD] f32;
// Yb out: [B,T,NH*HD] bf16.
// ---------------------------------------------------------------------------
__global__ __launch_bounds__(32) void attn_k(const __bf16* __restrict__ Qb,
                                             const __bf16* __restrict__ Kb,
                                             const __bf16* __restrict__ Vt,
                                             const float* __restrict__ Gf,
                                             const float* __restrict__ og,
                                             __bf16* __restrict__ Yb)
{
  __shared__ __align__(16) __bf16 pShm[16 * 32];   // P tile C-layout -> A-layout
  const int qt  = blockIdx.x;           // q tile (16 rows)
  const int bh  = blockIdx.y;           // b*NH + h
  const int b   = bh / NH_;
  const int h   = bh % NH_;
  const int lane = threadIdx.x;
  const int col  = lane & 15;
  const int hi   = lane >> 4;
  const int q0   = qt * 16;
  const float scale = 0.08838834764831845f;   // 1/sqrt(128)

  // Preload Q (16 x 128) as 4 A-fragments
  v16bf qf[4];
#pragma unroll
  for (int ds = 0; ds < 4; ++ds) {
    const __bf16* p = Qb + ((size_t)bh * T_ + q0 + col) * HD_ + ds*32 + hi*16;
    load_frag_A(qf[ds], p);
  }

  const v8f vz = {0.f,0.f,0.f,0.f,0.f,0.f,0.f,0.f};
  v8f acc[8];                 // O accumulator 16 x 128
  float mrow[8], lrow[8];
#pragma unroll
  for (int j = 0; j < 8; ++j) acc[j] = vz;
#pragma unroll
  for (int r = 0; r < 8; ++r) { mrow[r] = -1e30f; lrow[r] = 0.f; }

  const int nchunk = (q0 + 16 + 31) >> 5;       // 32-key chunks (causal)
  for (int kc = 0; kc < nchunk; ++kc) {
    const int k0 = kc * 32;
    v8f sl = vz, sh = vz;                       // scores cols 0..15 / 16..31
#pragma unroll
    for (int ds = 0; ds < 4; ++ds) {
      v16bf kfl, kfh;
      load_frag_B(kfl, Kb + ((size_t)bh*T_ + k0 + col)      * HD_ + ds*32 + hi*16);
      load_frag_B(kfh, Kb + ((size_t)bh*T_ + k0 + 16 + col) * HD_ + ds*32 + hi*16);
      sl = wmma_bf16(qf[ds], kfl, sl);
      sh = wmma_bf16(qf[ds], kfh, sh);
    }

    float cm[8], alpha[8];
#pragma unroll
    for (int r = 0; r < 8; ++r) {
      int qi = q0 + r + 8*hi;
      float s0 = sl[r] * scale; if (k0 + col      > qi) s0 = -1e30f;
      float s1 = sh[r] * scale; if (k0 + 16 + col > qi) s1 = -1e30f;
      sl[r] = s0; sh[r] = s1;
      cm[r] = fmaxf(s0, s1);
    }
#pragma unroll
    for (int r = 0; r < 8; ++r) {
#pragma unroll
      for (int m = 1; m < 16; m <<= 1)          // reduce within 16-lane half
        cm[r] = fmaxf(cm[r], __shfl_xor(cm[r], m, 32));
      float mn = fmaxf(mrow[r], cm[r]);
      alpha[r] = __expf(mrow[r] - mn);
      mrow[r] = mn;
    }
    float rs[8];
#pragma unroll
    for (int r = 0; r < 8; ++r) {
      float p0 = __expf(sl[r] - mrow[r]);
      float p1 = __expf(sh[r] - mrow[r]);
      sl[r] = p0; sh[r] = p1;
      rs[r] = p0 + p1;
    }
#pragma unroll
    for (int r = 0; r < 8; ++r) {
#pragma unroll
      for (int m = 1; m < 16; m <<= 1) rs[r] += __shfl_xor(rs[r], m, 32);
      lrow[r] = lrow[r] * alpha[r] + rs[r];
    }
#pragma unroll
    for (int j = 0; j < 8; ++j)
#pragma unroll
      for (int r = 0; r < 8; ++r) acc[j][r] *= alpha[r];

    // P (16x32) from C-layout to row-major bf16 in LDS, re-read as A-fragment.
    // Single wave: LDS ops are in-order per wave; waits keep scheduler honest.
#pragma unroll
    for (int r = 0; r < 8; ++r) {
      int row = r + 8*hi;
      pShm[row*32 + col]      = (__bf16)sl[r];
      pShm[row*32 + col + 16] = (__bf16)sh[r];
    }
    __builtin_amdgcn_wave_barrier();
    asm volatile("s_wait_dscnt 0" ::: "memory");
    v16bf pf;
    load_frag_A(pf, &pShm[col*32 + hi*16]);
    __builtin_amdgcn_wave_barrier();

#pragma unroll
    for (int j = 0; j < 8; ++j) {
      v16bf vf;   // B-operand: cols = d tile j, K rows = 32 key positions
      load_frag_B(vf, Vt + ((size_t)bh*HD_ + j*16 + col) * T_ + k0 + hi*16);
      acc[j] = wmma_bf16(pf, vf, acc[j]);
    }
  }

  // Epilogue: y = acc/l, RMS over d=128, * o_gamma * silu(gate); write bf16.
  float linv[8], rinv[8];
#pragma unroll
  for (int r = 0; r < 8; ++r) {
    linv[r] = 1.0f / lrow[r];
    float ssq = 0.f;
#pragma unroll
    for (int j = 0; j < 8; ++j) { float y = acc[j][r] * linv[r]; ssq += y*y; }
#pragma unroll
    for (int m = 1; m < 16; m <<= 1) ssq += __shfl_xor(ssq, m, 32);
    rinv[r] = rsqrtf(ssq * (1.0f/128.0f) + 1e-5f);
  }
#pragma unroll
  for (int r = 0; r < 8; ++r) {
    int qi = q0 + r + 8*hi;
    const float* grow = Gf + (((size_t)(b*T_ + qi)) * NH_ + h) * HD_;
    __bf16* yrow = Yb + ((size_t)(b*T_ + qi)) * (NH_*HD_) + h * HD_;
#pragma unroll
    for (int j = 0; j < 8; ++j) {
      int d = j*16 + col;
      float g = grow[d];
      float sig = g / (1.0f + __expf(-g));
      yrow[d] = (__bf16)(acc[j][r] * linv[r] * rinv[r] * og[d] * sig);
    }
  }
}

// ---------------------------------------------------------------------------
extern "C" void kernel_launch(void* const* d_in, const int* in_sizes, int n_in,
                              void* d_out, int out_size, void* d_ws, size_t ws_size,
                              hipStream_t stream)
{
  const float* x  = (const float*)d_in[0];
  const float* Wq = (const float*)d_in[1];
  const float* Wk = (const float*)d_in[2];
  const float* Wv = (const float*)d_in[3];
  const float* Wg = (const float*)d_in[4];
  const float* Wo = (const float*)d_in[5];
  const float* qg = (const float*)d_in[6];
  const float* kg = (const float*)d_in[7];
  const float* og = (const float*)d_in[8];
  float* out = (float*)d_out;

  char* w = (char*)d_ws;
  auto carve = [&](size_t bytes) -> char* {
    char* p = w; w += (bytes + 255) & ~(size_t)255; return p;
  };
  const size_t NE = (size_t)MT_ * H_;   // 8,388,608 elems
  const size_t WE = (size_t)H_ * H_;    // 4,194,304 elems

  __bf16* xb  = (__bf16*)carve(NE * 2);
  __bf16* Wqb = (__bf16*)carve(WE * 2);
  __bf16* Wkb = (__bf16*)carve(WE * 2);
  __bf16* Wvb = (__bf16*)carve(WE * 2);
  __bf16* Wgb = (__bf16*)carve(WE * 2);
  __bf16* Wob = (__bf16*)carve(WE * 2);
  float*  Qf  = (float*)carve(NE * 4);
  float*  Kf  = (float*)carve(NE * 4);
  float*  Vf  = (float*)carve(NE * 4);
  float*  Gf  = (float*)carve(NE * 4);
  __bf16* Qb2 = (__bf16*)carve(NE * 2);   // [B,NH,T,HD]
  __bf16* Kb2 = (__bf16*)carve(NE * 2);   // [B,NH,T,HD]
  __bf16* Vt  = (__bf16*)carve(NE * 2);   // [B,NH,HD,T]
  __bf16* Yb  = (__bf16*)carve(NE * 2);   // [B,T,NH*HD]

  // 1) bf16 conversions
  f32_to_bf16_k<<<4096, 256, 0, stream>>>(x,  xb,  (long)NE);
  f32_to_bf16_k<<<4096, 256, 0, stream>>>(Wq, Wqb, (long)WE);
  f32_to_bf16_k<<<4096, 256, 0, stream>>>(Wk, Wkb, (long)WE);
  f32_to_bf16_k<<<4096, 256, 0, stream>>>(Wv, Wvb, (long)WE);
  f32_to_bf16_k<<<4096, 256, 0, stream>>>(Wg, Wgb, (long)WE);
  f32_to_bf16_k<<<4096, 256, 0, stream>>>(Wo, Wob, (long)WE);

  // 2) QKVG projections: [4096,2048] = xb @ W^T
  dim3 gg(H_/128, MT_/128);   // (16, 32)
  gemm_bf16_nt<<<gg, 256, 0, stream>>>(xb, Wqb, Qf);
  gemm_bf16_nt<<<gg, 256, 0, stream>>>(xb, Wkb, Kf);
  gemm_bf16_nt<<<gg, 256, 0, stream>>>(xb, Wvb, Vf);
  gemm_bf16_nt<<<gg, 256, 0, stream>>>(xb, Wgb, Gf);

  // 3) RoPE + RMS (q,k) -> bf16 [B,NH,T,HD]
  rope_rms_k<<<MT_ * NH_, 32, 0, stream>>>(Qf, qg, Qb2);
  rope_rms_k<<<MT_ * NH_, 32, 0, stream>>>(Kf, kg, Kb2);

  // 4) V -> bf16 transposed [B,NH,HD,T]
  v_transpose_k<<<8192, 256, 0, stream>>>(Vf, Vt, (long)NE);

  // 5) flash attention + o-RMS + SiLU gate
  dim3 ag(T_/16, B_*NH_);     // (128, 32)
  attn_k<<<ag, 32, 0, stream>>>(Qb2, Kb2, Vt, Gf, og, Yb);

  // 6) output projection: out = Yb @ Wo^T (f32 out)
  gemm_bf16_nt<<<gg, 256, 0, stream>>>(Yb, Wob, out);
}
